// SpatialProcessor_60146722013279
// MI455X (gfx1250) — matrix operand order repaired and checked by reading, not compiled, not run on hardware
//
#include <hip/hip_runtime.h>
#include <hip/hip_bf16.h>

#define NN 4096          // NUM_NODES
#define HID 64
#define OUTD 32
#define NHEADS 4
#define MASKW (NN / 32)  // 128 u32 words per row
#define NEG_SLOPE 0.2f

typedef _Float16 v16h __attribute__((ext_vector_type(16)));
typedef float    v8f  __attribute__((ext_vector_type(8)));

#if __has_builtin(__builtin_amdgcn_wmma_f32_16x16x4_f32)
#define HAVE_WMMA_F32X4 1
typedef float v2f __attribute__((ext_vector_type(2)));
#endif

// ---- wave32 WMMA 16x16x32 f16 VGPR layout maps (CDNA5 ISA 7.12.2) ----
// A-matrix 16x32 f16: lane m=L%16 is row M; half=L/16 selects K sub-block.
__device__ __forceinline__ int kmapA(int v, int hf, int lo) {
    return ((v >> 2) << 4) + (hf << 3) + ((v & 3) << 1) + lo;
}
// B-matrix 32x16 f16, this code's order (not AMD's table): lanes 0-15 K=0..15, lanes 16-31 K=16..31.
// => element idx maps to K = hf*16 + idx  (contiguous!)
__device__ __forceinline__ int kmapB(int v, int hf, int lo) {
    return (hf << 4) + (v << 1) + lo;
}
// C/D 16x16 f32: lane holds (M = r + 8*half, N = lane%16) in VGPR r.

// ---------------------------------------------------------------
// Kernel 1: row-normalize embedding -> f32 + f16 copies
// ---------------------------------------------------------------
__global__ void k_normalize(const float* __restrict__ emb, float* __restrict__ ne32,
                            _Float16* __restrict__ ne16) {
    __shared__ float red[HID];
    int row = blockIdx.x;
    int c   = threadIdx.x;                // 64 threads
    float v = emb[row * HID + c];
    red[c] = v * v;
    __syncthreads();
    for (int s = 32; s > 0; s >>= 1) {
        if (c < s) red[c] += red[c + s];
        __syncthreads();
    }
    float inv = rsqrtf(red[0]);
    float nv  = v * inv;
    ne32[row * HID + c] = nv;
    ne16[row * HID + c] = (_Float16)nv;
}

// ---------------------------------------------------------------
// Kernel 2: adjacency mask via WMMA.  One wave computes a 16x32
// tile of ne@ne^T (two 16x16 C tiles), thresholds >0.5, packs bits
// with __ballot -> u32 mask words (no atomics, full coverage).
// Uses f32 16x16x4 WMMA when available (matches reference f32
// precision at the 0.5 threshold); falls back to f16 16x16x32.
// ---------------------------------------------------------------
__global__ __launch_bounds__(256)
void k_mask(const float* __restrict__ ne32, const _Float16* __restrict__ ne16,
            unsigned int* __restrict__ maskbits) {
    int wave = blockIdx.x * (blockDim.x >> 5) + (threadIdx.x >> 5);
    int lane = threadIdx.x & 31;
    const int tilesJ = NN / 32;           // 128
    int ti = wave / tilesJ;               // 0..255
    int tj = wave % tilesJ;               // 0..127
    int i0 = ti * 16, j0 = tj * 32;
    int m  = lane & 15;
    int hf = lane >> 4;

    v8f c0 = {}; v8f c1 = {};
#ifdef HAVE_WMMA_F32X4
    // f32 path: A 16x4 (lane row M=m, VGPR r -> K = 2*hf + r), B 4x16 symmetric.
    const float* arow  = ne32 + (i0 + m) * HID + 2 * hf;
    const float* b0row = ne32 + (j0 + m) * HID + 2 * hf;
    const float* b1row = ne32 + (j0 + 16 + m) * HID + 2 * hf;
    #pragma unroll
    for (int k0 = 0; k0 < HID; k0 += 4) {
        v2f a  = *(const v2f*)(arow + k0);
        v2f b0 = *(const v2f*)(b0row + k0);
        v2f b1 = *(const v2f*)(b1row + k0);
        c0 = __builtin_amdgcn_wmma_f32_16x16x4_f32(false, a, false, b0, (short)0, c0, false, false);
        c1 = __builtin_amdgcn_wmma_f32_16x16x4_f32(false, a, false, b1, (short)0, c1, false, false);
    }
#else
    #pragma unroll
    for (int k0 = 0; k0 < HID; k0 += 32) {
        v16h a, b0, b1;
        #pragma unroll
        for (int idx = 0; idx < 16; ++idx) {
            int KA = k0 + kmapA(idx >> 1, hf, idx & 1);
            int KB = k0 + kmapB(idx >> 1, hf, idx & 1);
            a[idx]  = ne16[(i0 + m) * HID + KA];
            b0[idx] = ne16[(j0 + m) * HID + KB];
            b1[idx] = ne16[(j0 + 16 + m) * HID + KB];
        }
        c0 = __builtin_amdgcn_wmma_f32_16x16x32_f16(false, a, false, b0, (short)0, c0, false, false);
        c1 = __builtin_amdgcn_wmma_f32_16x16x32_f16(false, a, false, b1, (short)0, c1, false, false);
    }
#endif
    #pragma unroll
    for (int r = 0; r < 8; ++r) {
        unsigned long long p0 = __ballot(c0[r] > 0.5f);
        unsigned long long p1 = __ballot(c1[r] > 0.5f);
        if (lane == 0) {
            unsigned lo = (unsigned)((p0 & 0xFFFFull) | ((p1 & 0xFFFFull) << 16));
            unsigned hi = (unsigned)(((p0 >> 16) & 0xFFFFull) | (((p1 >> 16) & 0xFFFFull) << 16));
            maskbits[(i0 + r)     * MASKW + tj] = lo;   // row i0+r
            maskbits[(i0 + r + 8) * MASKW + tj] = hi;   // row i0+r+8
        }
    }
}

// ---------------------------------------------------------------
// Kernel 3: input projection  h0 = x @ W_proj + b_proj  (16 -> 64)
// ---------------------------------------------------------------
__global__ void k_proj(const float* __restrict__ x, const float* __restrict__ Wp,
                       const float* __restrict__ bp, float* __restrict__ h0, int total) {
    int idx = blockIdx.x * blockDim.x + threadIdx.x;
    if (idx >= total) return;
    int c  = idx & (HID - 1);
    int bn = idx >> 6;
    const float* xr = x + bn * 16;
    float acc = bp[c];
    #pragma unroll
    for (int k = 0; k < 16; ++k) acc = fmaf(xr[k], Wp[k * HID + c], acc);
    h0[idx] = acc;
}

// ---------------------------------------------------------------
// Kernel 4: per-layer feature transform  hw = h_in @ W  (64 -> FOUT).
// Writes row-major f32 (for score dots) and a transposed, head-padded
// f16 tensor hwT[b][head][f(16)][N] so the WMMA B operand is a single
// contiguous 32B load per lane.
// ---------------------------------------------------------------
template <int FOUT>
__global__ void k_hw(const float* __restrict__ hin, const float* __restrict__ W,
                     float* __restrict__ hw32, _Float16* __restrict__ hwT, int total) {
    constexpr int F = FOUT / NHEADS;
    int idx = blockIdx.x * blockDim.x + threadIdx.x;
    if (idx >= total) return;
    int c  = idx % FOUT;
    int bn = idx / FOUT;
    int b  = bn / NN;
    int n  = bn - b * NN;
    const float* hr = hin + bn * HID;
    float acc = 0.f;
    #pragma unroll 8
    for (int k = 0; k < HID; ++k) acc = fmaf(hr[k], W[k * FOUT + c], acc);
    hw32[idx] = acc;
    int hd = c / F, f = c - hd * F;
    hwT[(((b * NHEADS + hd) * 16 + f) * NN) + n] = (_Float16)acc;
}

// ---------------------------------------------------------------
// Kernel 5: attention scores  s_src/s_dst[b,n,h] = <hw[b,n,h,:], a>
// ---------------------------------------------------------------
template <int F>
__global__ void k_scores(const float* __restrict__ hw32, const float* __restrict__ a_src,
                         const float* __restrict__ a_dst, float* __restrict__ ssrc,
                         float* __restrict__ sdst, int total) {
    int idx = blockIdx.x * blockDim.x + threadIdx.x;
    if (idx >= total) return;
    int hd = idx & (NHEADS - 1);
    int bn = idx >> 2;
    const float* hr = hw32 + bn * (NHEADS * F) + hd * F;
    float s1 = 0.f, s2 = 0.f;
    #pragma unroll
    for (int f = 0; f < F; ++f) {
        s1 = fmaf(hr[f], a_src[hd * F + f], s1);
        s2 = fmaf(hr[f], a_dst[hd * F + f], s2);
    }
    ssrc[idx] = s1;
    sdst[idx] = s2;
}

// ---------------------------------------------------------------
// Kernel 6: fused masked-softmax attention aggregation.
// Block = 8 waves sharing (b, head); each wave owns a 16-row i-tile.
// - s_src staged in LDS; block max folded into the load (softmax
//   shift uses the monotone-LeakyReLU upper bound lrelu(sd+max s_src),
//   exact under shift invariance -> no masked max pass needed).
// - Wave-uniform sparsity skip of all-zero 32-col mask words.
// - exp tile -> WMMA A; contiguous hwT slice -> WMMA B; denominator
//   accumulated per lane, reduced via shfl_xor(16) across K-halves.
// ---------------------------------------------------------------
template <int F, bool RELU>
__global__ __launch_bounds__(256)
void k_aggregate(const _Float16* __restrict__ hwT, const float* __restrict__ ssrc,
                 const float* __restrict__ sdst, const unsigned int* __restrict__ maskbits,
                 const float* __restrict__ bias, float* __restrict__ out) {
    __shared__ float s_src_sh[NN];     // 16 KB
    __shared__ float red_sh[256];
    __shared__ float l_sh[8 * 16];

    int g  = blockIdx.x & 31;          // i-tile group (8 tiles each)
    int hd = (blockIdx.x >> 5) & 3;    // head
    int b  = blockIdx.x >> 7;          // batch

    float lmax = -1e30f;
    for (int j = threadIdx.x; j < NN; j += blockDim.x) {
        float v = ssrc[(b * NN + j) * NHEADS + hd];
        s_src_sh[j] = v;
        lmax = fmaxf(lmax, v);
    }
    red_sh[threadIdx.x] = lmax;
    __syncthreads();
    for (int s = 128; s > 0; s >>= 1) {
        if (threadIdx.x < (unsigned)s)
            red_sh[threadIdx.x] = fmaxf(red_sh[threadIdx.x], red_sh[threadIdx.x + s]);
        __syncthreads();
    }
    float smax = red_sh[0];

    int wv   = threadIdx.x >> 5;
    int lane = threadIdx.x & 31;
    int m    = lane & 15;
    int hf   = lane >> 4;
    int i0   = (g * 8 + wv) * 16;
    int row  = i0 + m;

    float sd   = sdst[(b * NN + row) * NHEADS + hd];
    float emax = sd + smax;
    float mx   = emax > 0.f ? emax : NEG_SLOPE * emax;   // row upper bound
    const unsigned int* mrow  = maskbits + row * MASKW;
    const _Float16*     hwrow = hwT + (size_t)((b * NHEADS + hd) * 16 + m) * NN;

    v8f acc = {};
    float lsum = 0.f;
    for (int j0 = 0; j0 < NN; j0 += 32) {
        unsigned w = mrow[j0 >> 5];
        if (__ballot(w != 0) == 0ull) continue;          // sparsity skip (wave-uniform)
        if (j0 + 32 < NN)
            __builtin_prefetch(hwrow + j0 + 32 + (hf << 4), 0, 0);
        v16h a;
        #pragma unroll
        for (int idx = 0; idx < 16; ++idx) {
            int KA = (hf << 4) + idx;
            float p = 0.f;
            if ((w >> KA) & 1u) {
                float e = sd + s_src_sh[j0 + KA];
                e = e > 0.f ? e : NEG_SLOPE * e;
                p = __expf(e - mx);
            }
            _Float16 p16 = (_Float16)p;
            a[idx] = p16;
            lsum += (float)p16;
        }
        v16h bm = *(const v16h*)(hwrow + j0 + (hf << 4));  // contiguous 32B load
        acc = __builtin_amdgcn_wmma_f32_16x16x32_f16(false, a, false, bm, (short)0, acc, false, false);
    }
    lsum += __shfl_xor(lsum, 16, 32);
    if (hf == 0) l_sh[wv * 16 + m] = lsum;
    __syncthreads();

    // writeback: C lane holds (M = r + 8*hf, N = m)
    if (F == 16 || m < F) {
        #pragma unroll
        for (int r = 0; r < 8; ++r) {
            int orow = i0 + r + 8 * hf;
            float l  = l_sh[wv * 16 + r + 8 * hf];
            float v  = acc[r] / l + bias[hd * F + m];
            if (RELU) v = fmaxf(v, 0.f);
            out[(b * NN + orow) * (NHEADS * F) + hd * F + m] = v;
        }
    }
}

// ---------------------------------------------------------------
// Host-side orchestration
// ---------------------------------------------------------------
extern "C" void kernel_launch(void* const* d_in, const int* in_sizes, int n_in,
                              void* d_out, int out_size, void* d_ws, size_t ws_size,
                              hipStream_t stream) {
    const float* x   = (const float*)d_in[0];
    const float* emb = (const float*)d_in[1];
    const float* Wp  = (const float*)d_in[2];
    const float* bp  = (const float*)d_in[3];
    const float* W1  = (const float*)d_in[4];
    const float* a1s = (const float*)d_in[5];
    const float* a1d = (const float*)d_in[6];
    const float* b1  = (const float*)d_in[7];
    const float* W2  = (const float*)d_in[8];
    const float* a2s = (const float*)d_in[9];
    const float* a2d = (const float*)d_in[10];
    const float* b2  = (const float*)d_in[11];
    float* outp = (float*)d_out;
    char* ws = (char*)d_ws;

    const int B = 2;
    // workspace layout (all naturally 256B aligned)
    size_t OFF_NE32  = 0;                                       // 4096*64*4   = 1M
    size_t OFF_NE16  = OFF_NE32  + (size_t)NN * HID * 4;        // 4096*64*2   = 512K
    size_t OFF_MASK  = OFF_NE16  + (size_t)NN * HID * 2;        // 4096*128*4  = 2M
    size_t OFF_H0    = OFF_MASK  + (size_t)NN * MASKW * 4;      // 2*4096*64*4 = 2M
    size_t OFF_HW1   = OFF_H0    + (size_t)B * NN * HID * 4;    // 2M
    size_t OFF_HWT1  = OFF_HW1   + (size_t)B * NN * HID * 4;    // 2*4*16*4096*2 = 1M
    size_t OFF_S1S   = OFF_HWT1  + (size_t)B * NHEADS * 16 * NN * 2;
    size_t OFF_S1D   = OFF_S1S   + (size_t)B * NN * NHEADS * 4;
    size_t OFF_R1    = OFF_S1D   + (size_t)B * NN * NHEADS * 4; // 2M
    size_t OFF_HW2   = OFF_R1    + (size_t)B * NN * HID * 4;    // 1M
    size_t OFF_HWT2  = OFF_HW2   + (size_t)B * NN * OUTD * 4;   // 1M (padded)
    size_t OFF_S2S   = OFF_HWT2  + (size_t)B * NHEADS * 16 * NN * 2;
    size_t OFF_S2D   = OFF_S2S   + (size_t)B * NN * NHEADS * 4;

    float*    ne32  = (float*)(ws + OFF_NE32);
    _Float16* ne16  = (_Float16*)(ws + OFF_NE16);
    unsigned* maskb = (unsigned*)(ws + OFF_MASK);
    float*    h0    = (float*)(ws + OFF_H0);
    float*    hw1f  = (float*)(ws + OFF_HW1);
    _Float16* hwT1  = (_Float16*)(ws + OFF_HWT1);
    float*    s1s   = (float*)(ws + OFF_S1S);
    float*    s1d   = (float*)(ws + OFF_S1D);
    float*    r1    = (float*)(ws + OFF_R1);
    float*    hw2f  = (float*)(ws + OFF_HW2);
    _Float16* hwT2  = (_Float16*)(ws + OFF_HWT2);
    float*    s2s   = (float*)(ws + OFF_S2S);
    float*    s2d   = (float*)(ws + OFF_S2D);

    // 1) normalize embeddings
    k_normalize<<<NN, HID, 0, stream>>>(emb, ne32, ne16);

    // 2) adjacency bitmask: 256*128 = 32768 waves, 8 waves/block
    k_mask<<<(NN / 16) * (NN / 32) / 8, 256, 0, stream>>>(ne32, ne16, maskb);

    // 3) input projection
    {
        int total = B * NN * HID;
        k_proj<<<(total + 255) / 256, 256, 0, stream>>>(x, Wp, bp, h0, total);
    }

    // ---- layer 1 (F=16 per head, ReLU) ----
    {
        int total = B * NN * HID;
        k_hw<HID><<<(total + 255) / 256, 256, 0, stream>>>(h0, W1, hw1f, hwT1, total);
        int ts = B * NN * NHEADS;
        k_scores<16><<<(ts + 255) / 256, 256, 0, stream>>>(hw1f, a1s, a1d, s1s, s1d, ts);
        k_aggregate<16, true><<<B * NHEADS * 32, 256, 0, stream>>>(hwT1, s1s, s1d, maskb, b1, r1);
    }

    // ---- layer 2 (F=8 per head, no ReLU) -> d_out ----
    {
        // zero-fill padded transposed features (heads padded 8 -> 16 rows)
        hipMemsetAsync(hwT2, 0, (size_t)B * NHEADS * 16 * NN * 2, stream);
        int total = B * NN * OUTD;
        k_hw<OUTD><<<(total + 255) / 256, 256, 0, stream>>>(r1, W2, hw2f, hwT2, total);
        int ts = B * NN * NHEADS;
        k_scores<8><<<(ts + 255) / 256, 256, 0, stream>>>(hw2f, a2s, a2d, s2s, s2d, ts);
        k_aggregate<8, false><<<B * NHEADS * 32, 256, 0, stream>>>(hwT2, s2s, s2d, maskb, b2, outp);
    }
}
